// MultiheadAttention_71073118814586
// MI455X (gfx1250) — matrix operand order repaired and checked by reading, not compiled, not run on hardware
//
#include <hip/hip_runtime.h>
#include <hip/hip_bf16.h>
#include <stdint.h>

typedef __attribute__((ext_vector_type(16))) __bf16 v16bf;
typedef __attribute__((ext_vector_type(8)))  __bf16 v8bf;
typedef __attribute__((ext_vector_type(8)))  float  v8f;
typedef __attribute__((ext_vector_type(4)))  unsigned int v4u;
typedef __attribute__((ext_vector_type(8)))  int v8i;
typedef __attribute__((ext_vector_type(4)))  int v4i;

#if __has_builtin(__builtin_amdgcn_tensor_load_to_lds) && \
    __has_builtin(__builtin_amdgcn_s_wait_tensorcnt)
#define USE_TDM 1
#else
#define USE_TDM 0
#endif

#define WMMA_BF16(a, b, c) \
  __builtin_amdgcn_wmma_f32_16x16x32_bf16(false, (a), false, (b), (short)0, (c), false, false)

// ---- fragment load helpers (contiguous k order, same for A and B) ----------------

// A-matrix 16x32 bf16: lane m = lane%16, hf = lane/16.
// element e -> K = 16*hf + e (contiguous 32B, split as 2x16B)
__device__ inline v16bf load_a_frag(const __bf16* rowp, int hf) {
  v8bf lo = *(const v8bf*)(rowp + 16 * hf);
  v8bf hi = *(const v8bf*)(rowp + 16 * hf + 8);
  v16bf a;
#pragma unroll
  for (int i = 0; i < 8; ++i) { a[i] = lo[i]; a[8 + i] = hi[i]; }
  return a;
}

// B-matrix 32x16 bf16, read from an N-major (transposed) tile:
// lane n = lane%16, element e -> K = 16*hf + e (contiguous 32B, split as 2x16B)
__device__ inline v16bf load_b_frag(const __bf16* p) {
  v8bf lo = *(const v8bf*)(p);
  v8bf hi = *(const v8bf*)(p + 8);
  v16bf b;
#pragma unroll
  for (int i = 0; i < 8; ++i) { b[i] = lo[i]; b[8 + i] = hi[i]; }
  return b;
}

// ---- fp32 -> bf16 convert -------------------------------------------------

__global__ void __launch_bounds__(256) f32_to_bf16_kernel(const float* __restrict__ in,
                                                          __bf16* __restrict__ out, int n) {
  int i = blockIdx.x * blockDim.x + threadIdx.x;
  int stride = gridDim.x * blockDim.x;
  for (; i < n; i += stride) out[i] = (__bf16)in[i];
}

// ---- tiled WMMA GEMM: C[M,N] = A[M,K](bf16) * B[K,N](bf16) + bias --------
// MODE 0: write fp32 C.   MODE 1: scatter bf16 into Q/K/V [B,H,S,hd] layout.
// block = 256 threads (8 waves, 2x4), tile 128x128, K-step 32, double-buffered LDS.

template <int MODE>
__global__ void __launch_bounds__(256) gemm_bf16_kernel(
    const __bf16* __restrict__ A, const __bf16* __restrict__ B,
    const float* __restrict__ bias, float* __restrict__ outF,
    __bf16* __restrict__ outQ, __bf16* __restrict__ outK, __bf16* __restrict__ outV,
    int M, int N, int K) {
  const int LS = 40;  // padded LDS row stride (bf16 units)
  __shared__ __bf16 As[2][128 * 40];   // M-major, 128 rows x 32 K
  __shared__ __bf16 BsT[2][128 * 40];  // N-major, 128 cols x 32 K

  const int t = threadIdx.x;
  const int wave = t >> 5, lane = t & 31, hf = lane >> 4, ln = lane & 15;
  const int wm = wave >> 2, wn = wave & 3;
  const int m0 = blockIdx.y * 128, n0 = blockIdx.x * 128;

  const v8f vzero = {0.f, 0.f, 0.f, 0.f, 0.f, 0.f, 0.f, 0.f};
  v8f acc[4][2];
#pragma unroll
  for (int i = 0; i < 4; ++i)
#pragma unroll
    for (int j = 0; j < 2; ++j) acc[i][j] = vzero;

  // per-thread staging registers for one tile (A: 8x u32, B: 8x u32)
  uint32_t aReg[8], bReg[8];

  auto load_tile_regs = [&](int k0) {
#pragma unroll
    for (int i = 0; i < 8; ++i) {
      int p = i * 256 + t;
      int r = p >> 4, c2 = p & 15;
      aReg[i] = *(const uint32_t*)(A + (size_t)(m0 + r) * K + k0 + c2 * 2);
    }
#pragma unroll
    for (int i = 0; i < 8; ++i) {
      int p = i * 256 + t;
      int r = p >> 6, c2 = p & 63;
      bReg[i] = *(const uint32_t*)(B + (size_t)(k0 + r) * N + n0 + c2 * 2);
    }
  };
  auto store_tile_lds = [&](int buf) {
#pragma unroll
    for (int i = 0; i < 8; ++i) {
      int p = i * 256 + t;
      int r = p >> 4, c2 = p & 15;
      *(uint32_t*)&As[buf][r * LS + c2 * 2] = aReg[i];
    }
#pragma unroll
    for (int i = 0; i < 8; ++i) {
      int p = i * 256 + t;
      int r = p >> 6, c2 = p & 63;
      __bf16 b0 = ((const __bf16*)&bReg[i])[0];
      __bf16 b1 = ((const __bf16*)&bReg[i])[1];
      BsT[buf][(2 * c2) * LS + r] = b0;
      BsT[buf][(2 * c2 + 1) * LS + r] = b1;
    }
  };

  load_tile_regs(0);
  store_tile_lds(0);
  __syncthreads();

  int buf = 0;
  for (int k0 = 0; k0 < K; k0 += 32) {
    const bool more = (k0 + 32) < K;
    if (more) load_tile_regs(k0 + 32);  // global loads overlap with compute below

    // preload all fragments, then run the WMMA chain back-to-back
    v16bf af[4], bfr[2];
#pragma unroll
    for (int i = 0; i < 4; ++i)
      af[i] = load_a_frag(&As[buf][(wm * 64 + i * 16 + ln) * LS], hf);
#pragma unroll
    for (int j = 0; j < 2; ++j)
      bfr[j] = load_b_frag(&BsT[buf][(wn * 32 + j * 16 + ln) * LS + 16 * hf]);

#pragma unroll
    for (int i = 0; i < 4; ++i)
#pragma unroll
      for (int j = 0; j < 2; ++j)
        acc[i][j] = WMMA_BF16(af[i], bfr[j], acc[i][j]);

    if (more) store_tile_lds(buf ^ 1);
    __syncthreads();
    buf ^= 1;
  }

  // epilogue: C layout -> row = base + v + 8*hf, col = base + ln
#pragma unroll
  for (int i = 0; i < 4; ++i) {
#pragma unroll
    for (int j = 0; j < 2; ++j) {
      int n_g = n0 + wn * 32 + j * 16 + ln;
      float bv = bias[n_g];
#pragma unroll
      for (int v = 0; v < 8; ++v) {
        int m_g = m0 + wm * 64 + i * 16 + v + 8 * hf;
        float r = acc[i][j][v] + bv;
        if (MODE == 0) {
          outF[(size_t)m_g * N + n_g] = r;
        } else {
          int which = n_g >> 10, rem = n_g & 1023;
          int h = rem >> 6, d = rem & 63;
          int b = m_g >> 11, s = m_g & 2047;
          __bf16* dst = (which == 0) ? outQ : (which == 1) ? outK : outV;
          dst[(((size_t)b * 16 + h) * 2048 + s) * 64 + d] = (__bf16)r;
        }
      }
    }
  }
}

// ---- flash attention: causal softmax(Q K^T / 8) V ------------------------
// grid = (B*H, S/64), block = 128 (4 waves). Wave w owns 16 query rows.
// Q/K/V bf16 [B,H,S,64]; Z bf16 [B,S,D] (head-concatenated).
// K tile is fetched by the Tensor Data Mover (with LDS row padding applied by
// the TDM pad fields); V is loaded manually because it must be transposed.

__global__ void __launch_bounds__(128) attn_kernel(const __bf16* __restrict__ Qm,
                                                   const __bf16* __restrict__ Km,
                                                   const __bf16* __restrict__ Vm,
                                                   __bf16* __restrict__ Z) {
  const int S = 2048, HD = 64, PS = 72;  // PS = padded LDS stride
  __shared__ __bf16 Ks[64 * 72];   // key-major: Ks[key][d]
  __shared__ __bf16 VsT[64 * 72];  // d-major:   VsT[d][key]
  __shared__ __bf16 Ps[4 * 16 * 72];

  const int bh = blockIdx.x, qt = blockIdx.y;
  const int t = threadIdx.x;
  const int wave = t >> 5, lane = t & 31, hf = lane >> 4, ln = lane & 15;
  const int qrow0 = qt * 64 + wave * 16;
  const float scale = 0.125f;  // 1/sqrt(64)

  // load this wave's Q block as two A fragments (d 0..31, 32..63)
  const __bf16* qrow = Qm + ((size_t)bh * S + qrow0 + ln) * HD;
  v16bf qa0 = load_a_frag(qrow, hf);
  v16bf qa1 = load_a_frag(qrow + 32, hf);

  const v8f vzero = {0.f, 0.f, 0.f, 0.f, 0.f, 0.f, 0.f, 0.f};
  v8f zacc[4];
#pragma unroll
  for (int i = 0; i < 4; ++i) zacc[i] = vzero;
  float mrun[8], lrun[8];
#pragma unroll
  for (int v = 0; v < 8; ++v) { mrun[v] = -1e30f; lrun[v] = 0.f; }

  __bf16* ps = &Ps[wave * 16 * PS];
  const int nkv = (qt + 1) * 64;

  for (int j0 = 0; j0 < nkv; j0 += 64) {
    const __bf16* kp = Km + ((size_t)bh * S + j0) * HD;
    const __bf16* vp = Vm + ((size_t)bh * S + j0) * HD;

#if USE_TDM
    // K tile via Tensor Data Mover: 64x64 bf16 tile, row-major, with LDS
    // padding 32 DWORDs data + 4 DWORDs pad -> matches PS=72 stride.
    if (wave == 0) {
      uint64_t ga = (uint64_t)(uintptr_t)kp;
      uint32_t ldsa = (uint32_t)(uintptr_t)&Ks[0];
      v4u g0;
      g0[0] = 1u;                                   // count=1 (valid user D#)
      g0[1] = ldsa;                                 // lds_addr
      g0[2] = (uint32_t)ga;                         // global_addr[31:0]
      g0[3] = (uint32_t)((ga >> 32) & 0x01FFFFFFu)  // global_addr[56:32]
              | (2u << 30);                         // type=2 ("image")
      v8i g1;
      g1[0] = (int)((1u << 16)      // data_size = 2 bytes
                    | (1u << 20)    // pad_enable
                    | (4u << 22)    // pad_interval: 32 DWORDs
                    | (3u << 25));  // pad_amount: 4 DWORDs
      g1[1] = (int)(64u << 16);     // tensor_dim0 = 64 (low16)
      g1[2] = (int)(64u << 16);     // tensor_dim0 hi | tensor_dim1 = 64 (low16)
      g1[3] = (int)(64u << 16);     // tensor_dim1 hi | tile_dim0 = 64
      g1[4] = (int)(64u);           // tile_dim1 = 64, tile_dim2 = 0
      g1[5] = 64;                   // tensor_dim0_stride = 64 (low32)
      g1[6] = 0;
      g1[7] = 0;
      v4i z4 = {0, 0, 0, 0};
#if defined(__clang_major__) && (__clang_major__ >= 23)
      v8i z8 = {0, 0, 0, 0, 0, 0, 0, 0};
      __builtin_amdgcn_tensor_load_to_lds(g0, g1, z4, z4, z8, 0);
#else
      __builtin_amdgcn_tensor_load_to_lds(g0, g1, z4, z4, 0);
#endif
    }
#endif

    // V tile load (transposed into VsT); K tile too if no TDM available.
#pragma unroll
    for (int i = 0; i < 16; ++i) {
      int p = i * 128 + t;
      int r = p >> 5, c2 = p & 31;
#if !USE_TDM
      *(uint32_t*)&Ks[r * PS + c2 * 2] = *(const uint32_t*)(kp + r * HD + c2 * 2);
#endif
      uint32_t vv = *(const uint32_t*)(vp + r * HD + c2 * 2);
      VsT[(2 * c2) * PS + r] = ((const __bf16*)&vv)[0];
      VsT[(2 * c2 + 1) * PS + r] = ((const __bf16*)&vv)[1];
    }

    // prefetch next iteration's K/V tiles into cache
    if (j0 + 64 < nkv) {
      __builtin_prefetch(kp + 64 * HD + t * 32, 0, 0);
      __builtin_prefetch(vp + 64 * HD + t * 32, 0, 0);
    }

#if USE_TDM
    if (wave == 0) __builtin_amdgcn_s_wait_tensorcnt(0);
#endif
    __syncthreads();

    // preload all K fragments, then the score WMMA chain
    v16bf kbf[4][2];
#pragma unroll
    for (int jn = 0; jn < 4; ++jn) {
      kbf[jn][0] = load_b_frag(&Ks[(jn * 16 + ln) * PS + 16 * hf]);       // d 0..31
      kbf[jn][1] = load_b_frag(&Ks[(jn * 16 + ln) * PS + 32 + 16 * hf]);  // d 32..63
    }
    v8f sc[4];
#pragma unroll
    for (int jn = 0; jn < 4; ++jn) {
      v8f s = vzero;
      s = WMMA_BF16(qa0, kbf[jn][0], s);
      s = WMMA_BF16(qa1, kbf[jn][1], s);
      int col = j0 + jn * 16 + ln;
#pragma unroll
      for (int v = 0; v < 8; ++v) {
        int row = qrow0 + v + 8 * hf;
        s[v] = (col <= row) ? s[v] * scale : -1e30f;
      }
      sc[jn] = s;
    }

    // online softmax: row max across 16-lane half-group
    float mx[8];
#pragma unroll
    for (int v = 0; v < 8; ++v)
      mx[v] = fmaxf(fmaxf(sc[0][v], sc[1][v]), fmaxf(sc[2][v], sc[3][v]));
#pragma unroll
    for (int mask = 1; mask <= 8; mask <<= 1)
#pragma unroll
      for (int v = 0; v < 8; ++v) mx[v] = fmaxf(mx[v], __shfl_xor(mx[v], mask, 32));

    float sf[8];
#pragma unroll
    for (int v = 0; v < 8; ++v) {
      float mnew = fmaxf(mrun[v], mx[v]);
      sf[v] = __expf(mrun[v] - mnew);
      mrun[v] = mnew;
    }
    float rs[8];
#pragma unroll
    for (int v = 0; v < 8; ++v) rs[v] = 0.f;
#pragma unroll
    for (int jn = 0; jn < 4; ++jn)
#pragma unroll
      for (int v = 0; v < 8; ++v) {
        float p = __expf(sc[jn][v] - mrun[v]);
        sc[jn][v] = p;
        rs[v] += p;
      }
#pragma unroll
    for (int mask = 1; mask <= 8; mask <<= 1)
#pragma unroll
      for (int v = 0; v < 8; ++v) rs[v] += __shfl_xor(rs[v], mask, 32);
#pragma unroll
    for (int v = 0; v < 8; ++v) lrun[v] = lrun[v] * sf[v] + rs[v];
#pragma unroll
    for (int jh = 0; jh < 4; ++jh)
#pragma unroll
      for (int v = 0; v < 8; ++v) zacc[jh][v] *= sf[v];

    // transpose P (C layout) -> A-fragment layout via per-wave LDS
#pragma unroll
    for (int jn = 0; jn < 4; ++jn)
#pragma unroll
      for (int v = 0; v < 8; ++v)
        ps[(v + 8 * hf) * PS + jn * 16 + ln] = (__bf16)sc[jn][v];

    v16bf pa0 = load_a_frag(&ps[ln * PS], hf);       // keys 0..31
    v16bf pa1 = load_a_frag(&ps[ln * PS + 32], hf);  // keys 32..63

    // preload all V fragments, then the P*V WMMA chain
    v16bf vbf[4][2];
#pragma unroll
    for (int jh = 0; jh < 4; ++jh) {
      vbf[jh][0] = load_b_frag(&VsT[(jh * 16 + ln) * PS + 16 * hf]);       // keys 0..31
      vbf[jh][1] = load_b_frag(&VsT[(jh * 16 + ln) * PS + 32 + 16 * hf]);  // keys 32..63
    }
#pragma unroll
    for (int jh = 0; jh < 4; ++jh) {
      zacc[jh] = WMMA_BF16(pa0, vbf[jh][0], zacc[jh]);
      zacc[jh] = WMMA_BF16(pa1, vbf[jh][1], zacc[jh]);
    }
    __syncthreads();
  }

  // write normalized Z into [B,S,D]
  const int b = bh >> 4, h = bh & 15;
#pragma unroll
  for (int v = 0; v < 8; ++v) {
    float linv = 1.f / lrun[v];
    int srow = qrow0 + v + 8 * hf;
#pragma unroll
    for (int jh = 0; jh < 4; ++jh) {
      int d = jh * 16 + ln;
      Z[((size_t)b * S + srow) * 1024 + h * 64 + d] = (__bf16)(zacc[jh][v] * linv);
    }
  }
}

// ---- launch ---------------------------------------------------------------

extern "C" void kernel_launch(void* const* d_in, const int* in_sizes, int n_in,
                              void* d_out, int out_size, void* d_ws, size_t ws_size,
                              hipStream_t stream) {
  (void)in_sizes; (void)n_in; (void)out_size; (void)ws_size;
  const float* x     = (const float*)d_in[0];
  const float* w_qkv = (const float*)d_in[1];
  const float* b_qkv = (const float*)d_in[2];
  const float* w_out = (const float*)d_in[3];
  const float* b_out = (const float*)d_in[4];
  float* out = (float*)d_out;

  const int B = 2, S = 2048, D = 1024, H = 16;
  const size_t nx = (size_t)B * S * D;      // 4 Mi elements
  const size_t nwq = (size_t)D * 3 * D;     // 3 Mi
  const size_t nwo = (size_t)D * D;         // 1 Mi

  char* ws = (char*)d_ws;
  __bf16* xb  = (__bf16*)ws; ws += nx * 2;
  __bf16* wqb = (__bf16*)ws; ws += nwq * 2;
  __bf16* wob = (__bf16*)ws; ws += nwo * 2;
  __bf16* qb  = (__bf16*)ws; ws += nx * 2;
  __bf16* kb  = (__bf16*)ws; ws += nx * 2;
  __bf16* vb  = (__bf16*)ws; ws += nx * 2;
  __bf16* zb  = (__bf16*)ws; ws += nx * 2;

  f32_to_bf16_kernel<<<2048, 256, 0, stream>>>(x, xb, (int)nx);
  f32_to_bf16_kernel<<<2048, 256, 0, stream>>>(w_qkv, wqb, (int)nwq);
  f32_to_bf16_kernel<<<1024, 256, 0, stream>>>(w_out, wob, (int)nwo);

  dim3 g1(3 * D / 128, B * S / 128);  // 24 x 32
  gemm_bf16_kernel<1><<<g1, 256, 0, stream>>>(xb, wqb, b_qkv, nullptr, qb, kb, vb,
                                              B * S, 3 * D, D);

  dim3 g2(B * H, S / 64);  // 32 x 32
  attn_kernel<<<g2, 128, 0, stream>>>(qb, kb, vb, zb);

  dim3 g3(D / 128, B * S / 128);  // 8 x 32
  gemm_bf16_kernel<0><<<g3, 256, 0, stream>>>(zb, wob, b_out, out, nullptr, nullptr,
                                              nullptr, B * S, D, D);
}
